// ELossFN_29867202576454
// MI455X (gfx1250) — 8-bit matrix operand order repaired and checked by reading, compiled, not run on hardware
//
#include <hip/hip_runtime.h>
#include <stdint.h>

#define N_NODES 4096
#define N_CLASS 10
#define GAMMA 1.0f
#define PER 0.001f

typedef int v8i __attribute__((ext_vector_type(8)));
typedef int vint4 __attribute__((vector_size(16)));

#if __has_builtin(__builtin_amdgcn_global_load_async_to_lds_b128) && \
    __has_builtin(__builtin_amdgcn_s_wait_asynccnt)
#define USE_ASYNC_LDS 1
#else
#define USE_ASYNC_LDS 0
#endif

#define WS_T      0
#define WS_SUBF   512
#define WS_INTF   1024
#define WS_NCNT   1536
#define WS_CE     1600
#define WS_DEG    1664
#define WS_EPOS   18048
#define WS_EP     34432
#define WS_ADJ    198400

__global__ __launch_bounds__(256) void k_init(char* ws) {
    int* w = (int*)ws;
    for (int i = threadIdx.x; i < 416; i += 256) w[i] = 0;
}

__global__ __launch_bounds__(256) void k_pack(const int* __restrict__ adjI,
                                              unsigned char* __restrict__ adjU8) {
    size_t g = (size_t)blockIdx.x * 256 + threadIdx.x;
    int4 v = ((const int4*)adjI)[g];
    uchar4 o;
    o.x = (unsigned char)v.x; o.y = (unsigned char)v.y;
    o.z = (unsigned char)v.z; o.w = (unsigned char)v.w;
    ((uchar4*)adjU8)[g] = o;
}

__global__ __launch_bounds__(256) void k_node(const float* __restrict__ preds,
                                              const int* __restrict__ labels,
                                              const int* __restrict__ maskI,
                                              const unsigned char* __restrict__ adjU8,
                                              int* __restrict__ deg,
                                              float* __restrict__ epos,
                                              float* __restrict__ EP,
                                              int* __restrict__ Ncnt,
                                              float* __restrict__ ceAcc) {
    int p = blockIdx.x * 256 + threadIdx.x;
    if (p >= N_NODES) return;
    const uint4* row = (const uint4*)(adjU8 + (size_t)p * N_NODES);
    int s = 0;
    #pragma unroll 4
    for (int i = 0; i < 256; ++i) {
        uint4 v = row[i];
        unsigned a = v.x, b = v.y, c = v.z, d = v.w;
        s += (int)((a & 255u) + ((a >> 8) & 255u) + ((a >> 16) & 255u) + (a >> 24));
        s += (int)((b & 255u) + ((b >> 8) & 255u) + ((b >> 16) & 255u) + (b >> 24));
        s += (int)((c & 255u) + ((c >> 8) & 255u) + ((c >> 16) & 255u) + (c >> 24));
        s += (int)((d & 255u) + ((d >> 8) & 255u) + ((d >> 16) & 255u) + (d >> 24));
    }
    deg[p] = s;

    float pr[N_CLASS];
    float M = -1e30f;
    #pragma unroll
    for (int c = 0; c < N_CLASS; ++c) { pr[c] = preds[p * N_CLASS + c]; M = fmaxf(M, pr[c]); }
    float se = 0.f;
    #pragma unroll
    for (int c = 0; c < N_CLASS; ++c) {
        se += __expf(pr[c] - M);
        EP[p * N_CLASS + c] = __expf(pr[c]);
    }
    int lp = labels[p];
    float logp = pr[lp] - M - __logf(se);
    atomicAdd(ceAcc, -logp * (1.0f / (float)N_NODES));
    epos[p] = __expf(-GAMMA * pr[lp]);
    if (maskI[p]) atomicAdd(&Ncnt[lp], 1);
}

__global__ __launch_bounds__(256) void k_gemm(const unsigned char* __restrict__ adj,
                                              const int* __restrict__ labels,
                                              const int* __restrict__ maskI,
                                              const int* __restrict__ deg,
                                              const float* __restrict__ epos,
                                              const float* __restrict__ EP,
                                              float* __restrict__ Tg,
                                              unsigned* __restrict__ subflag,
                                              unsigned* __restrict__ interflag) {
    __shared__ __align__(16) unsigned char At[256][64];
    __shared__ __align__(16) unsigned char Bt[128][64];
    __shared__ float    Tl[8][112];
    __shared__ unsigned sfl[112];
    __shared__ unsigned ifl[112];

    const int tid  = threadIdx.x;
    const int wave = tid >> 5;
    const int lane = tid & 31;
    const int m16  = lane & 15;
    const int hi   = lane >> 4;
    const int rowbase = blockIdx.y * 256;
    const int colbase = blockIdx.x * 128;

    for (int i = tid; i < 8 * 112; i += 256) ((float*)Tl)[i] = 0.f;
    for (int i = tid; i < 112; i += 256) { sfl[i] = 0u; ifl[i] = 0u; }
    __syncthreads();

    v8i acc[2][8] = {};

    const int brow  = tid & 127;
    const int bhalf = tid >> 7;

    for (int k0 = 0; k0 < N_NODES; k0 += 64) {
#if USE_ASYNC_LDS
        {
            const unsigned char* ga = adj + (size_t)(rowbase + tid) * N_NODES + k0;
            #pragma unroll
            for (int c = 0; c < 4; ++c)
                __builtin_amdgcn_global_load_async_to_lds_b128(
                    (vint4*)(ga + c * 16), (vint4*)&At[tid][c * 16], 0, 0);
            const unsigned char* gb = adj + (size_t)(colbase + brow) * N_NODES + k0 + bhalf * 32;
            #pragma unroll
            for (int c = 0; c < 2; ++c)
                __builtin_amdgcn_global_load_async_to_lds_b128(
                    (vint4*)(gb + c * 16), (vint4*)&Bt[brow][bhalf * 32 + c * 16], 0, 0);
        }
        __builtin_amdgcn_s_wait_asynccnt(0);
#else
        {
            const uint4* sa = (const uint4*)(adj + (size_t)(rowbase + tid) * N_NODES + k0);
            uint4 a0 = sa[0], a1 = sa[1], a2 = sa[2], a3 = sa[3];
            *(uint4*)&At[tid][0]  = a0; *(uint4*)&At[tid][16] = a1;
            *(uint4*)&At[tid][32] = a2; *(uint4*)&At[tid][48] = a3;
            const uint4* sb = (const uint4*)(adj + (size_t)(colbase + brow) * N_NODES + k0 + bhalf * 32);
            uint4 b0 = sb[0], b1 = sb[1];
            *(uint4*)&Bt[brow][bhalf * 32]      = b0;
            *(uint4*)&Bt[brow][bhalf * 32 + 16] = b1;
        }
#endif
        __syncthreads();

        v8i a0, a1;
        #pragma unroll
        for (int sr = 0; sr < 2; ++sr) {
            const unsigned char* arow = &At[wave * 32 + sr * 16 + m16][0];
            uint4 c0 = *(const uint4*)(arow +      hi * 16);
            uint4 c1 = *(const uint4*)(arow + 32 + hi * 16);
            v8i a;
            a[0] = (int)c0.x; a[1] = (int)c0.y; a[2] = (int)c0.z; a[3] = (int)c0.w;
            a[4] = (int)c1.x; a[5] = (int)c1.y; a[6] = (int)c1.z; a[7] = (int)c1.w;
            if (sr == 0) a0 = a; else a1 = a;
        }
        #pragma unroll
        for (int j = 0; j < 8; ++j) {
            v8i b;
            uint4 d0 = *(const uint4*)&Bt[j * 16 + m16][      hi * 16];
            uint4 d1 = *(const uint4*)&Bt[j * 16 + m16][32 + hi * 16];
            b[0] = (int)d0.x; b[1] = (int)d0.y; b[2] = (int)d0.z; b[3] = (int)d0.w;
            b[4] = (int)d1.x; b[5] = (int)d1.y; b[6] = (int)d1.z; b[7] = (int)d1.w;
            acc[0][j] = __builtin_amdgcn_wmma_i32_16x16x64_iu8(
                false, a0, false, b, acc[0][j], false, false);
            acc[1][j] = __builtin_amdgcn_wmma_i32_16x16x64_iu8(
                false, a1, false, b, acc[1][j], false, false);
        }
        __syncthreads();
    }

    const float s1 = 0.7310585786300049f;
    #pragma unroll
    for (int sr = 0; sr < 2; ++sr) {
        int   plab[8], pmask[8], pdeg[8];
        float pepos[8];
        int   pbase = rowbase + wave * 32 + sr * 16 + hi * 8;
        #pragma unroll
        for (int r = 0; r < 8; ++r) {
            int p = pbase + r;
            plab[r] = labels[p]; pmask[r] = maskI[p];
            pdeg[r] = deg[p];    pepos[r] = epos[p];
        }
        #pragma unroll
        for (int j = 0; j < 8; ++j) {
            int q     = colbase + j * 16 + m16;
            int lq    = labels[q];
            int mq    = maskI[q];
            int diagq = adj[(size_t)q * N_NODES + q];
            #pragma unroll
            for (int r = 0; r < 8; ++r) {
                if (!(pmask[r] && mq)) continue;
                int p     = pbase + r;
                int inter = acc[sr][j][r];
                int apq   = adj[(size_t)p * N_NODES + q];
                int sub   = pdeg[r] - inter - (apq & (1 - diagq));
                int ci    = plab[r] * N_CLASS + lq;
                if (sub   > 0) atomicOr(&sfl[ci], 1u);
                if (inter > 0) atomicOr(&ifl[ci], 1u);
                float ratio = (1.f + s1 * (float)sub) / (1.f + s1 * (float)inter);
                float v     = 1.f - 1.f / (1.f + __expf(-ratio));
                float E     = pepos[r] * EP[q * N_CLASS + plab[r]];
                atomicAdd(&Tl[wave][ci], E * v);
            }
        }
    }
    __syncthreads();

    for (int i = tid; i < N_CLASS * N_CLASS; i += 256) {
        float s = 0.f;
        #pragma unroll
        for (int w2 = 0; w2 < 8; ++w2) s += Tl[w2][i];
        if (s != 0.f)  atomicAdd(&Tg[i], s);
        if (sfl[i])    atomicOr(&subflag[i], 1u);
        if (ifl[i])    atomicOr(&interflag[i], 1u);
    }
}

__global__ __launch_bounds__(128) void k_final(const float* __restrict__ Tg,
                                               const unsigned* __restrict__ subflag,
                                               const unsigned* __restrict__ interflag,
                                               const int* __restrict__ Ncnt,
                                               const float* __restrict__ ceAcc,
                                               float* __restrict__ out) {
    __shared__ float red[128];
    int t = threadIdx.x;
    float val = 0.f;
    if (t < N_CLASS * N_CLASS) {
        int i = t / N_CLASS, j = t % N_CLASS;
        if (i != j && subflag[t] && interflag[t]) {
            float ni = fmaxf((float)Ncnt[i], 1.f);
            float nj = fmaxf((float)Ncnt[j], 1.f);
            val = Tg[t] / (ni * nj);
        }
    }
    red[t] = val;
    __syncthreads();
    for (int s = 64; s > 0; s >>= 1) {
        if (t < s) red[t] += red[t + s];
        __syncthreads();
    }
    if (t == 0) out[0] = ceAcc[0] + PER * red[0];
}

extern "C" void kernel_launch(void* const* d_in, const int* in_sizes, int n_in,
                              void* d_out, int out_size, void* d_ws, size_t ws_size,
                              hipStream_t stream) {
    (void)in_sizes; (void)n_in; (void)out_size; (void)ws_size;
    const float* preds  = (const float*)d_in[0];
    const int*   labels = (const int*)d_in[1];
    const int*   maskI  = (const int*)d_in[2];
    const int*   adjI   = (const int*)d_in[3];

    char* ws = (char*)d_ws;
    float*         Tg      = (float*)(ws + WS_T);
    unsigned*      subfl   = (unsigned*)(ws + WS_SUBF);
    unsigned*      interfl = (unsigned*)(ws + WS_INTF);
    int*           Ncnt    = (int*)(ws + WS_NCNT);
    float*         ceAcc   = (float*)(ws + WS_CE);
    int*           deg     = (int*)(ws + WS_DEG);
    float*         epos    = (float*)(ws + WS_EPOS);
    float*         EP      = (float*)(ws + WS_EP);
    unsigned char* adjU8   = (unsigned char*)(ws + WS_ADJ);

    k_init<<<1, 256, 0, stream>>>(ws);
    k_pack<<<(N_NODES * (size_t)N_NODES) / 4 / 256, 256, 0, stream>>>(adjI, adjU8);
    k_node<<<N_NODES / 256, 256, 0, stream>>>(preds, labels, maskI, adjU8,
                                              deg, epos, EP, Ncnt, ceAcc);
    dim3 grid(N_NODES / 128, N_NODES / 256);
    k_gemm<<<grid, 256, 0, stream>>>(adjU8, labels, maskI, deg, epos, EP,
                                     Tg, subfl, interfl);
    k_final<<<1, 128, 0, stream>>>(Tg, subfl, interfl, Ncnt, ceAcc, (float*)d_out);
}
